// unit_vit_36687610642706
// MI455X (gfx1250) — compile-verified, matrix operand order checked by reading, not run on hardware
//
#include <hip/hip_runtime.h>
#include <hip/hip_bf16.h>

typedef __attribute__((ext_vector_type(16))) _Float16 v16h;
typedef __attribute__((ext_vector_type(8)))  _Float16 v8h;
typedef __attribute__((ext_vector_type(8)))  float    v8f;
typedef __attribute__((ext_vector_type(4)))  float    v4f;

#define NN   64
#define CC   256
#define TT   120
#define VV   25
#define VP   32
#define HH   8
#define HDD  32
#define KG   5

// workspace layout (bytes)
#define WS_WQKV 0
#define WS_PROJ (768 * 256 * 2)            // 393216
#define WS_RPE  (WS_PROJ + 256 * 256 * 2)  // 524288

// ---------------------------------------------------------------------------
// Weight prep: fuse q_w + kv_w into one [768,256] f16 matrix; f16 proj & rpe.
// ---------------------------------------------------------------------------
__global__ void prep_weights(const float* __restrict__ q_w,
                             const float* __restrict__ kv_w,
                             const float* __restrict__ proj_w,
                             const float* __restrict__ rpe,
                             _Float16* __restrict__ wqkv,
                             _Float16* __restrict__ projh,
                             _Float16* __restrict__ rpeh) {
    int i = blockIdx.x * blockDim.x + threadIdx.x;
    if (i < 768 * 256) {
        int r = i >> 8, c = i & 255;
        float v = (r < 256) ? q_w[r * 256 + c] : kv_w[(r - 256) * 256 + c];
        wqkv[i] = (_Float16)v;
    }
    if (i < 256 * 256) projh[i] = (_Float16)proj_w[i];
    if (i < 5 * 256)   rpeh[i]  = (_Float16)rpe[i];
}

// ---------------------------------------------------------------------------
// WMMA fragment helpers (CDNA5 16-bit A/B striping): lane L holds row
// r = L&15 of a row-major [16][K] source, k-chunks kb=(L>>4)*8:
// [kb..kb+7] and [kb+16..kb+23].  A source = [M][K], B source = [N][K]
// (D = A * Bsrc^T + C).
// ---------------------------------------------------------------------------
__device__ inline v16h load_frag(const _Float16* p, int ld, int lane) {
    int r  = lane & 15;
    int kb = (lane >> 4) << 3;
    const _Float16* row = p + r * ld + kb;
    v8h a = *(const v8h*)(row);
    v8h b = *(const v8h*)(row + 16);
    v16h f;
#pragma unroll
    for (int i = 0; i < 8; ++i) { f[i] = a[i]; f[i + 8] = b[i]; }
    return f;
}

__device__ inline v16h load_frag_rows(const _Float16* p, int ld, int lane, int nrows) {
    int r  = lane & 15;
    int kb = (lane >> 4) << 3;
    v16h f;
#pragma unroll
    for (int i = 0; i < 16; ++i) f[i] = (_Float16)0.0f;
    if (r < nrows) {
        const _Float16* row = p + r * ld + kb;
        v8h a = *(const v8h*)(row);
        v8h b = *(const v8h*)(row + 16);
#pragma unroll
        for (int i = 0; i < 8; ++i) { f[i] = a[i]; f[i + 8] = b[i]; }
    }
    return f;
}

__device__ inline v8f wmma16(v16h a, v16h b, v8f c) {
    return __builtin_amdgcn_wmma_f32_16x16x32_f16(false, a, false, b,
                                                  (short)0, c, false, false);
}

// ---------------------------------------------------------------------------
// Fused block: one workgroup (8 waves / 256 thr, wave32) per (n,t) tile.
// ---------------------------------------------------------------------------
__global__ __launch_bounds__(256) void fused_block(
    const float* __restrict__ x, const float* __restrict__ jl,
    const int* __restrict__ hops, const float* __restrict__ nw,
    const float* __restrict__ nb, const float* __restrict__ proj_b,
    const float* __restrict__ w1, const float* __restrict__ outer,
    const float* __restrict__ alpha, const float* __restrict__ pe_w,
    const _Float16* __restrict__ wqkv, const _Float16* __restrict__ projh,
    const _Float16* __restrict__ rpeh, float* __restrict__ out)
{
    extern __shared__ char smem[];
    _Float16* XN  = (_Float16*)(smem);            // [32][256] xn, v-major
    _Float16* OL  = (_Float16*)(smem);            // overlay: o^T, v-major
    _Float16* QL  = (_Float16*)(smem + 16384);    // [32][256] q, v-major
    _Float16* KL  = (_Float16*)(smem + 32768);    // [32][256] k, v-major
    _Float16* MIX = (_Float16*)(smem + 32768);    // overlay: [8][32][32]
    _Float16* VL  = (_Float16*)(smem + 49152);    // [256][32] v, dim-major
    _Float16* EV  = (_Float16*)(smem + 65536);    // [32][256] e, v-major
    char* TR = smem + 81920;                      // transient region (57344 B)
    float* XB   = (float*)(TR);                   // ph1: [256][32] x f32
    float* Zb   = (float*)(TR + 32768);           // ph1: [256][8]
    float* Z2   = (float*)(TR + 40960);           // ph1: [256][8]
    float* ATT  = (float*)(TR);                   // ph3: [8][32mk][36] col-major
    float* QR   = (float*)(TR + 36864);           // ph3: [8][16hop][32nq]
    float* BIAS = (float*)(TR + 53248);           // ph3: [8][32]
    float* PS   = (float*)(TR + 54272);           // ph1: [8][64] partial stats
    float* MS   = (float*)(TR + 56320);           // ph1: [32][2] mean/rstd
    float* CS   = (float*)(TR + 56576);           // ph1: [8] jl col sums

    const int tid  = threadIdx.x;
    const int lane = tid & 31;
    const int wave = __builtin_amdgcn_readfirstlane(tid >> 5);  // wave-uniform
    const int blk  = blockIdx.x;
    const int n    = blk / TT;
    const int t    = blk % TT;

    __builtin_prefetch(wqkv + tid * 64, 0, 0);    // warm L2 with weights

    const float* xbase = x + ((size_t)(n * CC) * TT + t) * VV;

    // ---- phase 1a: stage x tile (f32, V padded to 32) ----
#pragma unroll
    for (int i = 0; i < 32; ++i) {
        int idx = tid + i * 256;
        int c = idx >> 5, v = idx & 31;
        XB[idx] = (v < VV) ? xbase[(size_t)c * (TT * VV) + v] : 0.0f;
    }
    if (tid < KG) {
        float s = 0.f;
        for (int v = 0; v < VV; ++v) s += jl[v * KG + tid];
        CS[tid] = s;
    }
    __syncthreads();

    // ---- phase 1b: LayerNorm stats (segmented reduction) ----
    {
        float s = 0.f, s2 = 0.f;
        for (int j = 0; j < 32; ++j) {
            float u = XB[(wave * 32 + j) * 32 + lane];
            s += u; s2 += u * u;
        }
        PS[wave * 64 + lane]      = s;
        PS[wave * 64 + 32 + lane] = s2;
    }
    __syncthreads();
    if (tid < VP) {
        float s = 0.f, s2 = 0.f;
        for (int seg = 0; seg < 8; ++seg) {
            s  += PS[seg * 64 + tid];
            s2 += PS[seg * 64 + 32 + tid];
        }
        float mu  = s * (1.0f / CC);
        float var = s2 * (1.0f / CC) - mu * mu;
        MS[tid * 2]     = mu;
        MS[tid * 2 + 1] = rsqrtf(var + 1e-5f);
    }
    __syncthreads();

    // ---- phase 1c: xn -> f16, v-major, packed row stores ----
    {
        int v  = tid >> 3;
        int c0 = (tid & 7) * 32;
        float mu = MS[v * 2], rs = MS[v * 2 + 1];
        bool valid = (v < VV);
#pragma unroll
        for (int b = 0; b < 4; ++b) {
            v8h hv;
#pragma unroll
            for (int j = 0; j < 8; ++j) {
                int c = c0 + b * 8 + j;
                float val = valid ? (XB[c * 32 + v] - mu) * rs * nw[c] + nb[c] : 0.f;
                hv[j] = (_Float16)val;
            }
            *(v8h*)(XN + v * 256 + c0 + b * 8) = hv;
        }
    }
    // ---- phase 1d-1: z = x @ jl_norm ----
#pragma unroll
    for (int i = 0; i < 8; ++i) {
        int idx = tid + i * 256;
        int c = idx >> 3, kk = idx & 7;
        float s = 0.f;
        if (kk < KG) {
            for (int v = 0; v < VV; ++v) s += XB[c * 32 + v] * jl[v * KG + kk];
            s *= (1.0f / CS[kk]);
        }
        Zb[idx] = s;
    }
    __syncthreads();
    // ---- phase 1d-2: z2 = pe_w @ z ----
#pragma unroll
    for (int i = 0; i < 8; ++i) {
        int idx = tid + i * 256;
        int d = idx >> 3, kk = idx & 7;
        float s = 0.f;
        if (kk < KG) {
            const float* wr = pe_w + d * CC;
            for (int c = 0; c < CC; ++c) s += wr[c] * Zb[c * 8 + kk];
        }
        Z2[idx] = s;
    }
    __syncthreads();
    // ---- phase 1d-3: e = z2 @ jl^T -> f16, v-major, packed ----
    {
        int v  = tid >> 3;
        int d0 = (tid & 7) * 32;
        bool valid = (v < VV);
        float jv[KG];
#pragma unroll
        for (int kk = 0; kk < KG; ++kk) jv[kk] = valid ? jl[v * KG + kk] : 0.f;
#pragma unroll
        for (int b = 0; b < 4; ++b) {
            v8h hv;
#pragma unroll
            for (int j = 0; j < 8; ++j) {
                int d = d0 + b * 8 + j;
                float s = 0.f;
#pragma unroll
                for (int kk = 0; kk < KG; ++kk) s += Z2[d * 8 + kk] * jv[kk];
                hv[j] = (_Float16)s;
            }
            *(v8h*)(EV + v * 256 + d0 + b * 8) = hv;
        }
    }
    __syncthreads();

    // ---- phase 2: QKV = Wqkv[768,256] @ xn ----
    // Each wave owns M-tiles mt = wave + 8*it and computes BOTH N-tiles with
    // the same A fragments (halves L2 weight traffic vs per-(mt,nt) tiling).
    const int nl  = lane & 15;
    const int mbo = (lane >> 4) << 3;
    for (int it = 0; it < 6; ++it) {
        const int mt = __builtin_amdgcn_readfirstlane(wave + (it << 3));  // 0..47
        v8f acc0 = {}, acc1 = {};
        const _Float16* wbase = wqkv + (size_t)(mt * 16) * 256;
#pragma unroll
        for (int kk = 0; kk < 8; ++kk) {
            v16h a  = load_frag(wbase + kk * 32, 256, lane);
            v16h b0 = load_frag(XN + kk * 32, 256, lane);            // v 0..15
            v16h b1 = load_frag(XN + 16 * 256 + kk * 32, 256, lane); // v 16..31
            acc0 = wmma16(a, b0, acc0);
            acc1 = wmma16(a, b1, acc1);
        }
        int v0 = nl, v1 = 16 + nl;
        if (mt < 16) {              // Q rows: packed stores
            v8h h0, h1;
#pragma unroll
            for (int i = 0; i < 8; ++i) { h0[i] = (_Float16)acc0[i]; h1[i] = (_Float16)acc1[i]; }
            *(v8h*)(QL + v0 * 256 + mt * 16 + mbo) = h0;
            *(v8h*)(QL + v1 * 256 + mt * 16 + mbo) = h1;
        } else if (mt < 32) {       // K rows: packed stores
            v8h h0, h1;
#pragma unroll
            for (int i = 0; i < 8; ++i) { h0[i] = (_Float16)acc0[i]; h1[i] = (_Float16)acc1[i]; }
            *(v8h*)(KL + v0 * 256 + (mt - 16) * 16 + mbo) = h0;
            *(v8h*)(KL + v1 * 256 + (mt - 16) * 16 + mbo) = h1;
        } else {                    // V rows: dim-major (strided)
            int m0 = (mt - 32) * 16 + mbo;
#pragma unroll
            for (int i = 0; i < 8; ++i) {
                VL[(m0 + i) * 32 + v0] = (_Float16)acc0[i];
                VL[(m0 + i) * 32 + v1] = (_Float16)acc1[i];
            }
        }
    }
    __syncthreads();

    // ---- phase 3: attention scores (wave = head) ----
    {
        const int h = wave;
        v16h aq0 = load_frag(QL + 0 * 16 * 256 + h * HDD, 256, lane);
        v16h aq1 = load_frag(QL + 1 * 16 * 256 + h * HDD, 256, lane);
        v16h bk0 = load_frag(KL + 0 * 16 * 256 + h * HDD, 256, lane);
        v16h bk1 = load_frag(KL + 1 * 16 * 256 + h * HDD, 256, lane);
        v16h be0 = load_frag(EV + 0 * 16 * 256 + h * HDD, 256, lane);
        v16h be1 = load_frag(EV + 1 * 16 * 256 + h * HDD, 256, lane);
        v16h brp = load_frag_rows(rpeh + h * HDD, 256, lane, KG);
        v8f zero = {};
        v8f s00 = wmma16(aq0, bk0, zero); s00 = wmma16(aq0, be0, s00);
        v8f s01 = wmma16(aq0, bk1, zero); s01 = wmma16(aq0, be1, s01);
        v8f s10 = wmma16(aq1, bk0, zero); s10 = wmma16(aq1, be0, s10);
        v8f s11 = wmma16(aq1, bk1, zero); s11 = wmma16(aq1, be1, s11);
        v8f r0  = wmma16(aq0, brp, zero);
        v8f r1  = wmma16(aq1, brp, zero);

        float* attn_h = ATT + h * (32 * 36);   // [mk][nq] ld 36
        float* qr_h   = QR  + h * (16 * 32);   // [hop][nq] ld 32
        auto st_att = [&](v8f d, int mi, int ni) {
            float* p = attn_h + (ni * 16 + nl) * 36 + mi * 16 + mbo;
            v4f a, b;
#pragma unroll
            for (int j = 0; j < 4; ++j) { a[j] = d[j]; b[j] = d[4 + j]; }
            *(v4f*)p = a; *(v4f*)(p + 4) = b;
        };
        st_att(s00, 0, 0); st_att(s01, 0, 1);
        st_att(s10, 1, 0); st_att(s11, 1, 1);
        auto st_qr = [&](v8f d, int mi) {
            float* p = qr_h + nl * 32 + mi * 16 + mbo;
            v4f a, b;
#pragma unroll
            for (int j = 0; j < 4; ++j) { a[j] = d[j]; b[j] = d[4 + j]; }
            *(v4f*)p = a; *(v4f*)(p + 4) = b;
        };
        st_qr(r0, 0); st_qr(r1, 1);

        // group bias per key column: sum_c w1[h,c] * e_h[mk,c]
        float s = 0.f;
        const float* w1h = w1 + h * HDD;
        const v8h* ev8 = (const v8h*)(EV + lane * 256 + h * HDD);
#pragma unroll
        for (int b = 0; b < 4; ++b) {
            v8h e8 = ev8[b];
#pragma unroll
            for (int j = 0; j < 8; ++j) s += w1h[b * 8 + j] * (float)e8[j];
        }
        BIAS[h * 32 + lane] = s;
    }
    __syncthreads();

    // ---- phase 4: softmax + alpha-blend with `outer` (thread = head,row) ----
    {
        int h  = tid >> 5;
        int nq = tid & 31;
        _Float16* mixrow = MIX + h * (32 * 32) + nq * 32;
        _Float16 buf[32];
        if (nq < VV) {
            const float* attn_h = ATT + h * (32 * 36) + nq;   // stride 36 over mk
            const float* qr_h   = QR  + h * (16 * 32) + nq;   // stride 32 over hop
            const int* hrow = hops + nq * VV;
            const float scale = 0.1767766953f;  // 1/sqrt(HD)
            float vals[VV];
            float mx = -1e30f;
#pragma unroll
            for (int mk = 0; mk < VV; ++mk) {
                float val = (attn_h[mk * 36] + qr_h[hrow[mk] * 32] + BIAS[h * 32 + mk]) * scale;
                vals[mk] = val;
                mx = fmaxf(mx, val);
            }
            float ssum = 0.f;
#pragma unroll
            for (int mk = 0; mk < VV; ++mk) { vals[mk] = __expf(vals[mk] - mx); ssum += vals[mk]; }
            float inv = alpha[0] / ssum;
            const float* outr = outer + h * (VV * VV) + nq * VV;
#pragma unroll
            for (int mk = 0; mk < VV; ++mk) buf[mk] = (_Float16)(vals[mk] * inv + outr[mk]);
#pragma unroll
            for (int mk = VV; mk < 32; ++mk) buf[mk] = (_Float16)0.f;
        } else {
#pragma unroll
            for (int mk = 0; mk < 32; ++mk) buf[mk] = (_Float16)0.f;
        }
#pragma unroll
        for (int b = 0; b < 4; ++b) {
            v8h hv;
#pragma unroll
            for (int j = 0; j < 8; ++j) hv[j] = buf[b * 8 + j];
            *(v8h*)(mixrow + b * 8) = hv;
        }
    }
    __syncthreads();

    // ---- phase 5: o^T = v_h^T @ mix^T (wave = head), packed stores ----
    {
        const int h = wave;
        const _Float16* mixh = MIX + h * (32 * 32);
        v16h bm0 = load_frag(mixh, 32, lane);            // Bsrc rows nq 0..15
        v16h bm1 = load_frag(mixh + 16 * 32, 32, lane);  // Bsrc rows nq 16..31
        v16h av0 = load_frag(VL + (h * HDD + 0) * 32, 32, lane);   // A rows c 0..15
        v16h av1 = load_frag(VL + (h * HDD + 16) * 32, 32, lane);  // A rows c 16..31
        v8f zero = {};
        v8f d00 = wmma16(av0, bm0, zero);
        v8f d01 = wmma16(av0, bm1, zero);
        v8f d10 = wmma16(av1, bm0, zero);
        v8f d11 = wmma16(av1, bm1, zero);
        auto sto = [&](v8f d, int mi, int ni) {
            v8h hv;
#pragma unroll
            for (int i = 0; i < 8; ++i) hv[i] = (_Float16)d[i];
            *(v8h*)(OL + (ni * 16 + nl) * 256 + h * HDD + mi * 16 + mbo) = hv;
        };
        sto(d00, 0, 0); sto(d01, 0, 1);
        sto(d10, 1, 0); sto(d11, 1, 1);
    }
    __syncthreads();

    // ---- phase 6: out = proj_w @ o + proj_b + x ----
    // Same A-reuse scheme: each wave owns mt, computes both N-tiles.
    float* obase = out + ((size_t)(n * CC) * TT + t) * VV;
#pragma unroll
    for (int it = 0; it < 2; ++it) {
        const int mt = __builtin_amdgcn_readfirstlane(wave + (it << 3));  // 0..15
        v8f acc0 = {}, acc1 = {};
        const _Float16* wbase = projh + (size_t)(mt * 16) * 256;
#pragma unroll
        for (int kk = 0; kk < 8; ++kk) {
            v16h a  = load_frag(wbase + kk * 32, 256, lane);
            v16h b0 = load_frag(OL + kk * 32, 256, lane);
            v16h b1 = load_frag(OL + 16 * 256 + kk * 32, 256, lane);
            acc0 = wmma16(a, b0, acc0);
            acc1 = wmma16(a, b1, acc1);
        }
        int mb2 = mt * 16 + mbo;
        int v0 = nl, v1 = 16 + nl;
        {
#pragma unroll
            for (int i = 0; i < 8; ++i) {
                int m = mb2 + i;
                obase[(size_t)m * (TT * VV) + v0] =
                    acc0[i] + proj_b[m] + xbase[(size_t)m * (TT * VV) + v0];
            }
        }
        if (v1 < VV) {
#pragma unroll
            for (int i = 0; i < 8; ++i) {
                int m = mb2 + i;
                obase[(size_t)m * (TT * VV) + v1] =
                    acc1[i] + proj_b[m] + xbase[(size_t)m * (TT * VV) + v1];
            }
        }
    }
}

// ---------------------------------------------------------------------------
extern "C" void kernel_launch(void* const* d_in, const int* in_sizes, int n_in,
                              void* d_out, int out_size, void* d_ws, size_t ws_size,
                              hipStream_t stream) {
    const float* x      = (const float*)d_in[0];
    const float* jl     = (const float*)d_in[1];
    const int*   hops   = (const int*)d_in[2];
    const float* nw     = (const float*)d_in[3];
    const float* nb     = (const float*)d_in[4];
    const float* q_w    = (const float*)d_in[5];
    const float* kv_w   = (const float*)d_in[6];
    const float* proj_w = (const float*)d_in[7];
    const float* proj_b = (const float*)d_in[8];
    const float* rpe    = (const float*)d_in[9];
    const float* w1     = (const float*)d_in[10];
    const float* outer  = (const float*)d_in[11];
    const float* alpha  = (const float*)d_in[12];
    const float* pe_w   = (const float*)d_in[13];
    float* out = (float*)d_out;

    _Float16* wqkv  = (_Float16*)((char*)d_ws + WS_WQKV);
    _Float16* projh = (_Float16*)((char*)d_ws + WS_PROJ);
    _Float16* rpeh  = (_Float16*)((char*)d_ws + WS_RPE);

    prep_weights<<<768, 256, 0, stream>>>(q_w, kv_w, proj_w, rpe, wqkv, projh, rpeh);

    const size_t lds_bytes = 81920 + 57344;  // persistent + transient = 139264
    fused_block<<<NN * TT, 256, lds_bytes, stream>>>(
        x, jl, hops, nw, nb, proj_b, w1, outer, alpha, pe_w,
        wqkv, projh, rpeh, out);
}
